// ConvCapsule_89300960018599
// MI455X (gfx1250) — matrix operand order repaired and checked by reading, not compiled, not run on hardware
//
#include <hip/hip_runtime.h>

// ---------------- problem constants ----------------
#define IN_DIM   32
#define OUT_DIM  32
#define IN_ATOMS 8
#define OUT_ATOMS 8
#define KSZ      5
#define NROUTE   3
#define BATCH    8
#define HH       32
#define WW       32

#define KREAL  200          // 25 taps * 8 input atoms
#define KPAD   224          // padded to 7 * 32 for bf16 WMMA k-steps
#define NDIM   256          // 32 out caps * 8 atoms
#define PIX    4            // pixels per workgroup
#define MDIM   128          // PIX * IN_DIM
#define A_STRIDE 116        // dwords per A row in LDS (112 used, padded for banks)
#define V_STRIDE 258        // f32 per votes row in LDS (256 used, padded for banks)

typedef __attribute__((ext_vector_type(16))) __bf16        v16bf;
typedef __attribute__((ext_vector_type(8)))  float         v8f;
typedef __attribute__((ext_vector_type(4)))  unsigned int  u32x4;

__device__ __forceinline__ unsigned short f32_to_bf16_rne(float f) {
  unsigned int u = __float_as_uint(f);
  unsigned int r = 0x7FFFu + ((u >> 16) & 1u);
  return (unsigned short)((u + r) >> 16);
}

// ---------------- prepack: conv_w [5,5,8,256] f32 -> Bt[oc=256][k=224] bf16 (K-major rows)
__global__ void prep_w_kernel(const float* __restrict__ w, unsigned short* __restrict__ Bt) {
  int idx = blockIdx.x * blockDim.x + threadIdx.x;
  if (idx >= NDIM * KPAD) return;
  int oc = idx / KPAD;
  int kp = idx - oc * KPAD;                      // kp = tap*8 + ic
  float v = (kp < KREAL) ? w[kp * NDIM + oc] : 0.0f;
  Bt[oc * KPAD + kp] = f32_to_bf16_rne(v);
}

// ---------------- prepack: x [256,8,32,32] f32 -> xp [n][y][x][ic] bf16 (ic innermost)
__global__ void prep_x_kernel(const float* __restrict__ x, unsigned short* __restrict__ xp) {
  int idx = blockIdx.x * blockDim.x + threadIdx.x;   // 2,097,152 threads exactly
  int ic = idx & 7;
  int t  = idx >> 3;
  int xx = t & 31; t >>= 5;
  int yy = t & 31;
  int n  = t >> 5;
  xp[idx] = f32_to_bf16_rne(x[((n * IN_ATOMS + ic) * HH + yy) * WW + xx]);
}

// ---------------- fused conv(votes)-GEMM + 3x dynamic routing, one 4-pixel tile per WG
__launch_bounds__(512, 1)
__global__ void capsule_fused_kernel(const unsigned short* __restrict__ xp,
                                     const unsigned short* __restrict__ Bt,
                                     const float* __restrict__ conv_b,
                                     const float* __restrict__ biases,
                                     float* __restrict__ out) {
  extern __shared__ char smem[];
  unsigned int* Alds  = (unsigned int*)smem;                              // 128 x 116 dwords
  float* votes  = (float*)(smem + MDIM * A_STRIDE * 4);                   // 128 x 258 f32
  float* route  = votes + MDIM * V_STRIDE;                                // 128 x 32
  float* logits = route + MDIM * 32;                                      // 128 x 32
  float* preact = logits + MDIM * 32;                                     // 1024
  float* act    = preact + 1024;                                          // 1024

  const int tid = threadIdx.x;
  const int b   = blockIdx.x >> 8;            // 8 batches * 256 pixel-tiles
  const int pix = (blockIdx.x & 255) * PIX;
  const int y0  = pix >> 5;
  const int x0  = pix & 31;                   // multiple of 4

  const int wave = tid >> 5, lane = tid & 31;
  const int half = lane >> 4, l16 = lane & 15;
  const int mblk = (wave & 3) << 5;           // 4 M-blocks of 32
  const int nblk = (wave >> 2) << 6;          // 4 N-blocks of 64

  // light L2 prefetch of this wave's weight panel
  for (int r = lane; r < 64; r += 32)
    __builtin_prefetch(Bt + (nblk + r) * KPAD, 0, 1);

  // ---- stage A (implicit im2col, bf16, zero-padded) into LDS ----
  // A[m][kp]: m = p*32 + i, kp = (ky*5+kx)*8 + ic; stored as dword pairs (ic even/odd)
  for (int t = tid; t < MDIM * 112; t += 512) {
    int m  = t / 112, kd = t - m * 112;       // kd = kp/2
    int p  = m >> 5, i = m & 31;
    int tap = kd >> 2, icp = kd & 3;
    unsigned int v = 0u;
    if (tap < 25) {
      int ky = tap / 5, kx = tap - ky * 5;
      int yy = y0 + ky - 2;
      int xx = x0 + p + kx - 2;
      if ((unsigned)yy < 32u && (unsigned)xx < 32u) {
        int n = b * IN_DIM + i;
        v = ((const unsigned int*)xp)[((n * 32 + yy) * 32 + xx) * 4 + icp];
      }
    }
    Alds[m * A_STRIDE + kd] = v;
  }
  __syncthreads();

  // ---- GEMM: votes[128 x 256] = A[128 x 224] * B[224 x 256], bf16 WMMA f32 acc ----
  v8f acc[2][4];
#pragma unroll
  for (int ms = 0; ms < 2; ++ms)
#pragma unroll
    for (int ns = 0; ns < 4; ++ns)
#pragma unroll
      for (int e = 0; e < 8; ++e) acc[ms][ns][e] = 0.0f;

#pragma unroll
  for (int k0 = 0; k0 < KPAD; k0 += 32) {
    // A fragments (16x32 bf16): lanes 0-15 hold M rows, K 0..15; lanes 16-31 K 16..31
    v16bf afrag[2];
#pragma unroll
    for (int ms = 0; ms < 2; ++ms) {
      const unsigned int* abase =
          Alds + (mblk + ms * 16 + l16) * A_STRIDE + (k0 >> 1) + half * 8;
      union { u32x4 q[2]; v16bf v; } au;
      au.q[0] = *(const u32x4*)(abase);
      au.q[1] = *(const u32x4*)(abase + 4);
      afrag[ms] = au.v;
    }
    // B fragments (32x16 bf16): lane -> column N, lane half selects K 0..15 / 16..31
#pragma unroll
    for (int ns = 0; ns < 4; ++ns) {
      const unsigned short* bptr =
          Bt + (nblk + ns * 16 + l16) * KPAD + k0 + half * 16;
      union { u32x4 q[2]; v16bf v; } bu;
      bu.q[0] = *(const u32x4*)(bptr);
      bu.q[1] = *(const u32x4*)(bptr + 8);
#pragma unroll
      for (int ms = 0; ms < 2; ++ms) {
        acc[ms][ns] = __builtin_amdgcn_wmma_f32_16x16x32_bf16(
            false, afrag[ms], false, bu.v, (short)0, acc[ms][ns], false, false);
      }
    }
  }

  // ---- spill C tiles to LDS votes[m][oc] (+ conv bias) ----
#pragma unroll
  for (int ns = 0; ns < 4; ++ns) {
    int N = nblk + ns * 16 + l16;
    float cb = conv_b[N];
#pragma unroll
    for (int ms = 0; ms < 2; ++ms) {
      int Mb = mblk + ms * 16 + half * 8;     // C layout: lanes 0-15 M=r, lanes 16-31 M=8+r
#pragma unroll
      for (int r = 0; r < 8; ++r)
        votes[(Mb + r) * V_STRIDE + N] = acc[ms][ns][r] + cb;
    }
  }

  // init logits = 0 (the 'leak' logit is constant 0 in the reference)
  for (int t = tid; t < MDIM * 32; t += 512) logits[t] = 0.0f;
  __syncthreads();

  // ---- 3 routing iterations, all in LDS ----
  for (int it = 0; it < NROUTE; ++it) {
    // 1) leaky softmax over o per (p,i) row: denom includes exp(0 - max)
    if (tid < MDIM) {
      const float* lrow = logits + tid * 32;
      float mx = 0.0f;                         // leak logit = 0
      for (int o = 0; o < 32; ++o) mx = fmaxf(mx, lrow[o]);
      float denom = __expf(-mx);
      for (int o = 0; o < 32; ++o) denom += __expf(lrow[o] - mx);
      float inv = 1.0f / denom;
      float* rrow = route + tid * 32;
      for (int o = 0; o < 32; ++o) rrow[o] = __expf(lrow[o] - mx) * inv;
    }
    __syncthreads();

    // 2) preact[p][o][a] = sum_i votes[p,i,o*8+a] * route[p,i,o] + biases[o][a]
    for (int e = tid; e < PIX * NDIM; e += 512) {
      int p = e >> 8, oc = e & 255, o = oc >> 3;
      float s = biases[oc];
      const float* vb = votes + (p * 32) * V_STRIDE + oc;
      const float* rb = route + (p * 32) * 32 + o;
      for (int i = 0; i < 32; ++i)
        s = fmaf(vb[i * V_STRIDE], rb[i * 32], s);
      preact[e] = s;
    }
    __syncthreads();

    // 3) squash over atoms: act = s * norm / (1 + norm^2)
    if (tid < PIX * OUT_DIM) {
      const float* pr = preact + tid * 8;
      float ns = 0.0f;
      for (int a = 0; a < 8; ++a) ns = fmaf(pr[a], pr[a], ns);
      float scale = __fsqrt_rn(ns) / (1.0f + ns);
      float* ab = act + tid * 8;
      for (int a = 0; a < 8; ++a) ab[a] = pr[a] * scale;
    }
    __syncthreads();

    // 4) logits[p][i][o] += sum_a votes[p,i,o*8+a] * act[p,o,a]
    if (it < NROUTE - 1) {
      for (int e = tid; e < MDIM * 32; e += 512) {
        int p = e >> 10, i = (e >> 5) & 31, o = e & 31;
        const float* vb = votes + (p * 32 + i) * V_STRIDE + o * 8;
        const float* ab = act + p * 256 + o * 8;
        float s = 0.0f;
        for (int a = 0; a < 8; ++a) s = fmaf(vb[a], ab[a], s);
        logits[e] += s;
      }
      __syncthreads();
    }
  }

  // ---- write act -> out[b, o, y, x, a] flattened [8, 32*32*32, 8] ----
  for (int e = tid; e < PIX * NDIM; e += 512) {
    int p = e >> 8, o = (e >> 3) & 31, a = e & 7;
    out[(((b * OUT_DIM + o) * HH + y0) * WW + (x0 + p)) * OUT_ATOMS + a] = act[e];
  }
}

extern "C" void kernel_launch(void* const* d_in, const int* in_sizes, int n_in,
                              void* d_out, int out_size, void* d_ws, size_t ws_size,
                              hipStream_t stream) {
  const float* x      = (const float*)d_in[0];  // [256, 8, 32, 32]
  const float* conv_w = (const float*)d_in[1];  // [5, 5, 8, 256]
  const float* conv_b = (const float*)d_in[2];  // [256]
  const float* biases = (const float*)d_in[3];  // [32, 8, 1, 1]
  float* out = (float*)d_out;                   // [8, 32768, 8]

  unsigned short* Bt = (unsigned short*)d_ws;                     // 256*224 bf16 = 112 KB
  unsigned short* xp = (unsigned short*)((char*)d_ws + 131072);   // 2M bf16 = 4 MB

  prep_w_kernel<<<(NDIM * KPAD + 255) / 256, 256, 0, stream>>>(conv_w, Bt);
  prep_x_kernel<<<(BATCH * IN_DIM * IN_ATOMS * HH * WW) / 256, 256, 0, stream>>>(x, xp);

  const size_t lds_bytes =
      (size_t)MDIM * A_STRIDE * 4 +     // A staging        (59,392)
      (size_t)MDIM * V_STRIDE * 4 +     // votes            (132,096)
      (size_t)MDIM * 32 * 4 * 2 +       // route + logits   (32,768)
      1024 * 4 * 2;                     // preact + act     (8,192)  => 232,448 B

  capsule_fused_kernel<<<BATCH * (HH * WW / PIX), 512, lds_bytes, stream>>>(
      xp, Bt, conv_b, biases, out);
}
